// ElmoCRF_73263552135523
// MI455X (gfx1250) — matrix operand order repaired and checked by reading, not compiled, not run on hardware
//
#include <hip/hip_runtime.h>
#include <hip/hip_bf16.h>

typedef __attribute__((ext_vector_type(16))) _Float16 v16h;
typedef __attribute__((ext_vector_type(8)))  float    v8f;

#define BATCH 64
#define TLEN  512
#define DDIM  1024
#define LTAG  32
#define MROWS (BATCH * TLEN)          // 32768

// ---------------------------------------------------------------------------
// Kernel 1: emissions = emb[M,1024] x W^T[1024,32] + bias, via WMMA f16->f32.
// One wave => one 16x32 output tile (two 16x16 WMMA accumulators).
// 256 threads = 8 waves per block, 256 blocks => 2048 M-tiles.
// W (128KB f32) is transposed+converted once per block into LDS as f16 [k][n].
// ---------------------------------------------------------------------------
__global__ __launch_bounds__(256)
void emis_wmma_kernel(const float* __restrict__ A,   // [M, 1024]
                      const float* __restrict__ W,   // [32, 1024]
                      const float* __restrict__ bias,// [32]
                      float* __restrict__ E)         // [M, 32]
{
    __shared__ _Float16 Wt[DDIM * LTAG];             // 64 KB: Wt[k*32 + n] = W[n][k]

    const int tid = threadIdx.x;
    // Cooperative transpose: coalesced global reads of W, f32->f16 into LDS.
    for (int idx = tid; idx < DDIM * LTAG; idx += 256) {
        int n = idx >> 10;          // W row   (0..31)
        int k = idx & 1023;         // W col   (0..1023)
        Wt[k * LTAG + n] = (_Float16)W[idx];
    }
    __syncthreads();

    const int wave  = tid >> 5;
    const int lane  = tid & 31;
    const int mtile = blockIdx.x * 8 + wave;         // 0..2047
    const int rowbase = mtile * 16;

    // A fragment addressing (16-bit A 16x32 layout):
    //   lane holds row M = lane&15; K chunks [koff..koff+7] and [16+koff..16+koff+7]
    const int row  = lane & 15;
    const int koff = (lane >> 4) * 8;
    const float* ap = A + (size_t)(rowbase + row) * DDIM + koff;

    // B fragment addressing (16-bit B 32x16 layout): lane holds column N = lane&15, K in the same chunks as A.
    const _Float16* bp = &Wt[lane * LTAG];

    v8f acc0 = {};   // N = 0..15
    v8f acc1 = {};   // N = 16..31

    for (int kb = 0; kb < DDIM; kb += 32) {
        // prefetch the A stream ~4 iterations ahead (global_prefetch_b8)
        __builtin_prefetch(ap + kb + 128, 0, 0);

        const float4* p0 = (const float4*)(ap + kb);
        const float4* p1 = (const float4*)(ap + kb + 16);
        float4 a0 = p0[0], a1 = p0[1];
        float4 a2 = p1[0], a3 = p1[1];

        float tmp[16] = { a0.x, a0.y, a0.z, a0.w,  a1.x, a1.y, a1.z, a1.w,
                          a2.x, a2.y, a2.z, a2.w,  a3.x, a3.y, a3.z, a3.w };
        v16h af;
#pragma unroll
        for (int i = 0; i < 16; ++i) af[i] = (_Float16)tmp[i];

        // B fragments: column n = lane&15 (+16 for b1), K in table order matching A.
        v16h b0, b1;
#pragma unroll
        for (int i = 0; i < 16; ++i) {
            const int kk = kb + koff + i + (i >= 8 ? 8 : 0);
            b0[i] = Wt[kk * LTAG + row];
            b1[i] = Wt[kk * LTAG + row + 16];
        }

        acc0 = __builtin_amdgcn_wmma_f32_16x16x32_f16(false, af, false, b0,
                                                      (short)0, acc0, false, false);
        acc1 = __builtin_amdgcn_wmma_f32_16x16x32_f16(false, af, false, b1,
                                                      (short)0, acc1, false, false);
    }

    // C/D layout: lane 0-15 -> N=lane, M = r + 0; lane 16-31 -> N=lane-16, M = r + 8.
    const int ncol = lane & 15;
    const int mofs = (lane >> 4) * 8;
    const float bv0 = bias[ncol];
    const float bv1 = bias[ncol + 16];
#pragma unroll
    for (int r = 0; r < 8; ++r) {
        const size_t m = (size_t)(rowbase + mofs + r);
        E[m * LTAG + ncol]      = acc0[r] + bv0;
        E[m * LTAG + ncol + 16] = acc1[r] + bv1;
    }
}

// ---------------------------------------------------------------------------
// Kernel 2: fused CRF numerator + forward-algorithm denominator.
// One wave32 per batch; lane j owns alpha[j] and trans[:,j] in registers.
// mask is all-true in this instance.
// ---------------------------------------------------------------------------
__global__ __launch_bounds__(32)
void crf_kernel(const float* __restrict__ E,          // [B*T, 32]
                const int*   __restrict__ labels,     // [B, T]
                const float* __restrict__ start_trans,// [32]
                const float* __restrict__ end_trans,  // [32]
                const float* __restrict__ trans,      // [32, 32]
                float* __restrict__ partial)          // [B] : den - num
{
    const int b = blockIdx.x;
    const int j = threadIdx.x;                        // lane = tag index

    float tcol[LTAG];                                 // trans[i][j], i = 0..31
#pragma unroll
    for (int i = 0; i < LTAG; ++i) tcol[i] = trans[i * LTAG + j];

    const float* eb = E + (size_t)b * TLEN * LTAG;

    // ---- denominator: forward scan ----
    float alpha = start_trans[j] + eb[j];
    for (int t = 1; t < TLEN; ++t) {
        const float e = eb[t * LTAG + j];
        float s[LTAG];
        float m = -INFINITY;
#pragma unroll
        for (int i = 0; i < LTAG; ++i) {
            float ai = __shfl(alpha, i, 32);          // broadcast alpha[i]
            s[i] = ai + tcol[i];
            m = fmaxf(m, s[i]);
        }
        float sum = 0.0f;
#pragma unroll
        for (int i = 0; i < LTAG; ++i) sum += __expf(s[i] - m);
        alpha = m + __logf(sum) + e;
    }
    float v = alpha + end_trans[j];
    float m = v;
#pragma unroll
    for (int o = 16; o > 0; o >>= 1) m = fmaxf(m, __shfl_xor(m, o, 32));
    float sum = __expf(v - m);
#pragma unroll
    for (int o = 16; o > 0; o >>= 1) sum += __shfl_xor(sum, o, 32);
    const float den = m + __logf(sum);

    // ---- numerator: gold-path score, t striped across lanes ----
    const int* lb = labels + b * TLEN;
    float num = 0.0f;
    for (int t = j; t < TLEN; t += 32) {
        const int yt = lb[t];
        float c;
        if (t == 0) c = start_trans[yt] + eb[yt];
        else        c = trans[lb[t - 1] * LTAG + yt] + eb[t * LTAG + yt];
        num += c;
    }
#pragma unroll
    for (int o = 16; o > 0; o >>= 1) num += __shfl_xor(num, o, 32);
    num += end_trans[lb[TLEN - 1]];

    if (j == 0) partial[b] = den - num;               // = -llh_b
}

// ---------------------------------------------------------------------------
// Kernel 3: final mean over batch -> -mean(llh)
// ---------------------------------------------------------------------------
__global__ __launch_bounds__(32)
void reduce_kernel(const float* __restrict__ partial, float* __restrict__ out)
{
    const int j = threadIdx.x;
    float v = partial[j] + partial[j + 32];
#pragma unroll
    for (int o = 16; o > 0; o >>= 1) v += __shfl_xor(v, o, 32);
    if (j == 0) out[0] = v * (1.0f / (float)BATCH);
}

// ---------------------------------------------------------------------------
extern "C" void kernel_launch(void* const* d_in, const int* in_sizes, int n_in,
                              void* d_out, int out_size, void* d_ws, size_t ws_size,
                              hipStream_t stream)
{
    const float* emb    = (const float*)d_in[0];   // [B, T, D] f32
    const float* W      = (const float*)d_in[1];   // [L, D]    f32
    const float* bias   = (const float*)d_in[2];   // [L]       f32
    const float* stt    = (const float*)d_in[3];   // [L]       f32
    const float* ent    = (const float*)d_in[4];   // [L]       f32
    const float* trans  = (const float*)d_in[5];   // [L, L]    f32
    const int*   labels = (const int*)d_in[6];     // [B, T]    i32
    // d_in[7] = mask, all-true in this instance (torchcrf requirement) -> ignored

    float* E       = (float*)d_ws;                                   // 4 MB emissions
    float* partial = (float*)((char*)d_ws + (size_t)MROWS * LTAG * sizeof(float));

    // GEMM: 2048 M-tiles, 8 waves/block -> 256 blocks of 256 threads
    emis_wmma_kernel<<<256, 256, 0, stream>>>(emb, W, bias, E);

    // CRF: one wave32 per batch
    crf_kernel<<<BATCH, 32, 0, stream>>>(E, labels, stt, ent, trans, partial);

    // final scalar
    reduce_kernel<<<1, 32, 0, stream>>>(partial, (float*)d_out);
}
